// Conv2d_77489799955262
// MI455X (gfx1250) — matrix operand order repaired and checked by reading, not compiled, not run on hardware
//
#include <hip/hip_runtime.h>

// ---- problem constants (from reference) ----
#define BATCH   32
#define CIN     256
#define COUT    256
#define HH      56
#define WW      56
#define HWSZ    (HH*WW)           // 3136
#define NTOT    (BATCH*HWSZ)      // 100352
#define KTOT    (CIN*9)           // 2304
// ---- tiling ----
#define MTILE   256               // whole Cout per block -> each x gathered once
#define NTILE   128
#define KTILE   32
#define LDSK    (KTILE + 8)       // padded stride (80B) -> conflict-free b128 reads

typedef __attribute__((ext_vector_type(16))) _Float16     v16h;
typedef __attribute__((ext_vector_type(8)))  float        v8f;
typedef __attribute__((ext_vector_type(4)))  unsigned int u32x4;

// CDNA5 async global->LDS path (ASYNCcnt), guarded so compilation can't regress.
#if defined(__has_builtin)
#if __has_builtin(__builtin_amdgcn_global_load_async_to_lds_b128) && \
    __has_builtin(__builtin_amdgcn_s_wait_asynccnt)
#define USE_ASYNC_LDS 1
#endif
#endif
#ifndef USE_ASYNC_LDS
#define USE_ASYNC_LDS 0
#endif

#if USE_ASYNC_LDS
// Builtin parameter type per hipcc diagnostic: 'int __vector_size__(4*sizeof(int)) *'
typedef int v4i_vs __attribute__((vector_size(16)));
typedef __attribute__((address_space(1))) v4i_vs* async_gptr_t;  // global src
typedef __attribute__((address_space(3))) v4i_vs* async_lptr_t;  // LDS dst
#endif

// ------------------------------------------------------------------
// Prep 1: ternarize weights  T = (w_pos>0) - (w_neg>0)  -> f16
// ------------------------------------------------------------------
__global__ void ternarize_kernel(const float* __restrict__ wp,
                                 const float* __restrict__ wn,
                                 _Float16* __restrict__ wt, int n) {
    int i = blockIdx.x * blockDim.x + threadIdx.x;
    if (i < n) {
        float t = (wp[i] > 0.0f ? 1.0f : 0.0f) - (wn[i] > 0.0f ? 1.0f : 0.0f);
        wt[i] = (_Float16)t;   // exactly representable
    }
}

// ------------------------------------------------------------------
// Prep 2: activations f32 -> f16 (4 elements/thread)
// ------------------------------------------------------------------
__global__ void cvt_f16_kernel(const float* __restrict__ x,
                               _Float16* __restrict__ xh, int n4) {
    int i = blockIdx.x * blockDim.x + threadIdx.x;
    if (i < n4) {
        float4 v = ((const float4*)x)[i];
        union { _Float16 h[4]; unsigned long long u; } u;
        u.h[0] = (_Float16)v.x; u.h[1] = (_Float16)v.y;
        u.h[2] = (_Float16)v.z; u.h[3] = (_Float16)v.w;
        *(unsigned long long*)&xh[(size_t)i * 4] = u.u;
    }
}

// ------------------------------------------------------------------
// Main: implicit-GEMM ternary conv via V_WMMA_F32_16X16X32_F16
//   C[M=cout][N=b*3136+hw] = sum_K  W[M][K] * im2col(X)[K][N]
// 256 threads = 8 wave32 (4 Mwaves x 2 Nwaves); wave tile 64x64 ->
// 256x128 block tile, 16 WMMAs per K-step.
// Double-buffered pipeline: async weight DMA + register-staged im2col
// gather for tile t+1 overlap the 16 WMMAs of tile t; one barrier/step.
// ------------------------------------------------------------------
__global__ __launch_bounds__(256)
void conv_wmma_kernel(const _Float16* __restrict__ wt,   // [COUT][KTOT]
                      const _Float16* __restrict__ xh,   // [B][CIN][H][W]
                      float* __restrict__ out)           // [B][COUT][H][W]
{
    __shared__ __align__(16) _Float16 sA[2][MTILE * LDSK];  // [m][k] padded
    __shared__ __align__(16) _Float16 sB[2][NTILE * LDSK];  // [n][k] padded

    const int tid   = threadIdx.x;
    const int lane  = tid & 31;
    const int wv    = tid >> 5;
    const int wm    = wv & 3;            // 0..3  -> 64 couts each
    const int wn    = wv >> 2;           // 0..1  -> 64 cols each
    const int lrow  = lane & 15;
    const int khalf = lane >> 4;

    const int nbase = blockIdx.x * NTILE;

    // ---- im2col staging roles: this thread stages k'=kkq for 16 columns ----
    const int kkq = tid & 31;            // k index within K-tile
    const int ng  = tid >> 5;            // which group of 16 columns
    int paddr[16];                       // b*CIN*HWSZ + oh*56 + ow
    int pohw[16];                        // (oh<<8)|ow for bounds checks
#pragma unroll
    for (int i = 0; i < 16; ++i) {
        int npos = nbase + ng * 16 + i;  // < 100352 always (784*128 exact)
        int b  = npos / HWSZ;
        int hw = npos - b * HWSZ;
        int oh = hw / WW;
        int ow = hw - oh * WW;
        paddr[i] = b * (CIN * HWSZ) + oh * WW + ow;
        pohw[i]  = (oh << 8) | ow;
    }

    // ---------------- prologue: stage K-tile 0 into buffer 0 ----------------
#pragma unroll
    for (int j = 0; j < 4; ++j) {
        const _Float16* gsrc = wt + (size_t)tid * KTOT + j * 8;
        _Float16*       ldst = &sA[0][tid * LDSK + j * 8];
#if USE_ASYNC_LDS
        __builtin_amdgcn_global_load_async_to_lds_b128(
            (async_gptr_t)gsrc, (async_lptr_t)ldst, 0, 0);
#else
        *(u32x4*)ldst = *(const u32x4*)gsrc;
#endif
    }
    {
        int cin = kkq / 9;
        int r   = kkq - cin * 9;
        int kh  = r / 3;
        int kw  = r - kh * 3;
        int off = cin * HWSZ + (kh - 1) * WW + (kw - 1);
#pragma unroll
        for (int i = 0; i < 16; ++i) {
            int ih = (pohw[i] >> 8) + kh - 1;
            int iw = (pohw[i] & 0xff) + kw - 1;
            bool ok = (unsigned)ih < (unsigned)HH && (unsigned)iw < (unsigned)WW;
            _Float16 v = xh[ok ? paddr[i] + off : 0];
            sB[0][(ng * 16 + i) * LDSK + kkq] = ok ? v : (_Float16)0.0f;
        }
    }
#if USE_ASYNC_LDS
    __builtin_amdgcn_s_wait_asynccnt(0);
#endif
    __syncthreads();

    v8f acc[4][4] = {};
    int cur = 0;

    for (int kb = 0; kb < KTOT; kb += KTILE) {
        const int  kn   = kb + KTILE;
        const bool hasn = (kn < KTOT);

        // ---- 1) issue next tile's async weight DMA (no wait) ----
        unsigned okmask = 0;
        _Float16 breg[16];
        if (hasn) {
#pragma unroll
            for (int j = 0; j < 4; ++j) {
                const _Float16* gsrc = wt + (size_t)tid * KTOT + kn + j * 8;
                _Float16*       ldst = &sA[cur ^ 1][tid * LDSK + j * 8];
#if USE_ASYNC_LDS
                __builtin_amdgcn_global_load_async_to_lds_b128(
                    (async_gptr_t)gsrc, (async_lptr_t)ldst, 0, 0);
#else
                *(u32x4*)ldst = *(const u32x4*)gsrc;
#endif
            }
            if (kn + KTILE < KTOT)   // warm L2 for the tile after next
                __builtin_prefetch(wt + (size_t)tid * KTOT + kn + KTILE, 0, 0);

            // ---- 2) issue next tile's im2col gather into registers ----
            int kk  = kn + kkq;
            int cin = kk / 9;
            int r   = kk - cin * 9;
            int kh  = r / 3;
            int kw  = r - kh * 3;
            int off = cin * HWSZ + (kh - 1) * WW + (kw - 1);
#pragma unroll
            for (int i = 0; i < 16; ++i) {
                int ih = (pohw[i] >> 8) + kh - 1;
                int iw = (pohw[i] & 0xff) + kw - 1;
                bool ok = (unsigned)ih < (unsigned)HH && (unsigned)iw < (unsigned)WW;
                breg[i] = xh[ok ? paddr[i] + off : 0];   // clamped, select deferred
                okmask |= (unsigned)ok << i;
            }
        }

        // ---- 3) 16 WMMAs on current buffer (hides gather latency) ----
        union Frag { v16h v; u32x4 u[2]; };
        Frag bfr[4];
#pragma unroll
        for (int nt = 0; nt < 4; ++nt) {
            int ncol = wn * 64 + nt * 16 + lrow;
            // B 32x16: VGPR0-3 hold K = khalf*8..+7, VGPR4-7 hold K = 16+khalf*8..+7
            const _Float16* bp = &sB[cur][ncol * LDSK + khalf * 8];
            bfr[nt].u[0] = *(const u32x4*)(bp);
            bfr[nt].u[1] = *(const u32x4*)(bp + 16);
        }
#pragma unroll
        for (int mt = 0; mt < 4; ++mt) {
            int mrow = wm * 64 + mt * 16 + lrow;
            // A 16x32: VGPR0-3 hold K = khalf*8..+7, VGPR4-7 hold K = 16+khalf*8..+7
            Frag af;
            const _Float16* ap = &sA[cur][mrow * LDSK + khalf * 8];
            af.u[0] = *(const u32x4*)(ap);
            af.u[1] = *(const u32x4*)(ap + 16);
#pragma unroll
            for (int nt = 0; nt < 4; ++nt)
                acc[mt][nt] = __builtin_amdgcn_wmma_f32_16x16x32_f16(
                    false, af.v, false, bfr[nt].v,
                    (short)0, acc[mt][nt], false, false);
        }

        // ---- 4) commit gathered registers to the other buffer ----
        if (hasn) {
#pragma unroll
            for (int i = 0; i < 16; ++i)
                sB[cur ^ 1][(ng * 16 + i) * LDSK + kkq] =
                    ((okmask >> i) & 1u) ? breg[i] : (_Float16)0.0f;
        }

        // ---- 5) close the pipeline stage ----
#if USE_ASYNC_LDS
        __builtin_amdgcn_s_wait_asynccnt(0);
#endif
        __syncthreads();
        cur ^= 1;
    }

    // ---- store C: f32 per ISA C/D layout (VGPR r: M = r + khalf*8, N = lrow) ----
#pragma unroll
    for (int nt = 0; nt < 4; ++nt) {
        int npos = nbase + wn * 64 + nt * 16 + lrow;
        int b  = npos / HWSZ;
        int hw = npos - b * HWSZ;
        float* obase = out + (size_t)b * (COUT * HWSZ) + hw;
#pragma unroll
        for (int mt = 0; mt < 4; ++mt) {
            int cbase = wm * 64 + mt * 16 + khalf * 8;
#pragma unroll
            for (int r = 0; r < 8; ++r)
                obase[(size_t)(cbase + r) * HWSZ] = acc[mt][nt][r];
        }
    }
}

// ------------------------------------------------------------------
extern "C" void kernel_launch(void* const* d_in, const int* in_sizes, int n_in,
                              void* d_out, int out_size, void* d_ws, size_t ws_size,
                              hipStream_t stream) {
    const float* x  = (const float*)d_in[0];   // [32,256,56,56]
    const float* wp = (const float*)d_in[1];   // [256,256,3,3]
    const float* wn = (const float*)d_in[2];   // [256,256,3,3]
    float* out = (float*)d_out;                // [32,256,56,56]

    _Float16* wt = (_Float16*)d_ws;                      // 589824 halves
    _Float16* xh = wt + (size_t)COUT * KTOT;             // 25690112 halves

    int nw = COUT * KTOT;
    ternarize_kernel<<<(nw + 255) / 256, 256, 0, stream>>>(wp, wn, wt, nw);

    int n4 = (BATCH * CIN * HWSZ) / 4;
    cvt_f16_kernel<<<(n4 + 255) / 256, 256, 0, stream>>>(x, xh, n4);

    dim3 grid(NTOT / NTILE);   // 784 blocks, 256 threads (8 wave32)
    conv_wmma_kernel<<<grid, 256, 0, stream>>>(wt, xh, out);
}
